// DGQP_48146583388480
// MI455X (gfx1250) — matrix operand order repaired and checked by reading, not compiled, not run on hardware
//
#include <hip/hip_runtime.h>
#include <hip/hip_bf16.h>

typedef __attribute__((ext_vector_type(16))) _Float16 v16h;
typedef __attribute__((ext_vector_type(8)))  float    v8f;

#define HWSZ 25600   // 160*160
#define CCH  32      // channels
#define KDIM 20      // 4 groups * (top4 + mean)

// compare-exchange for descending sort
#define CE(a, b)                                   \
  {                                                \
    float _hi = fmaxf(e[a], e[b]);                 \
    float _lo = fminf(e[a], e[b]);                 \
    e[a] = _hi; e[b] = _lo;                        \
  }

// ReLU in one instruction: v_med3_f32(h, 0, +inf)
__device__ __forceinline__ float relu1(float h) {
    return __builtin_amdgcn_fmed3f(h, 0.0f, __builtin_inff());
}

// In-row (16-lane) butterfly add via DPP16: v += dpp(v)
template <int CTRL>
__device__ __forceinline__ float row_dpp_add(float v) {
    int iv = __float_as_int(v);
    int sh = __builtin_amdgcn_update_dpp(0, iv, CTRL, 0xF, 0xF, true);
    return v + __int_as_float(sh);
}
#define DPP_XOR1 0xB1   // quad_perm [1,0,3,2]
#define DPP_XOR2 0x4E   // quad_perm [2,3,0,1]
#define DPP_XOR4 0x141  // row_half_mirror
#define DPP_XOR8 0x140  // row_mirror

union A16 { v16h v; unsigned int d[8]; };

__global__ __launch_bounds__(256)
void dgqp_kernel(const float* __restrict__ x,
                 const float* __restrict__ w1,
                 const float* __restrict__ b1,
                 const float* __restrict__ w2,
                 const float* __restrict__ b2,
                 float* __restrict__ out,
                 int totalPix)
{
    const int lane   = threadIdx.x & 31;
    const int laneM  = lane & 15;
    const bool hiH   = (lane >= 16);

    // ---- Build B tiles (W1^T, 32x16 f16 per tile, 4 N-tiles) ----------------
    // 16-bit B 32x16 layout: lanes 0-15 hold K=0..7,16..23 (2 per VGPR), lanes 16-31
    // hold K=8..15,24..31. N = lane%16. K>=20 zero-padded.
    v16h B[4];
#pragma unroll
    for (int t = 0; t < 4; ++t) {
        const int n = t * 16 + laneM;
        v16h bv;
#pragma unroll
        for (int j = 0; j < 16; ++j) {
            const int k = (j < 8) ? ((hiH ? 8 : 0) + j) : (16 + (hiH ? 8 : 0) + (j - 8));
            const float wv = (k < KDIM) ? w1[n * KDIM + k] : 0.0f;
            bv[j] = (_Float16)wv;
        }
        B[t] = bv;
    }
    float w2v[4], b1v[4];
#pragma unroll
    for (int t = 0; t < 4; ++t) {
        w2v[t] = w2[t * 16 + laneM];
        b1v[t] = b1[t * 16 + laneM];
    }
    const float bias2 = b2[0];

    const int waveId = (blockIdx.x * blockDim.x + threadIdx.x) >> 5;
    const int nWaves = (gridDim.x * blockDim.x) >> 5;

    for (int base = waveId * 32; base < totalPix; base += nWaves * 32) {
        const int p    = base + lane;
        const int nimg = p / HWSZ;
        const int hw   = p - nimg * HWSZ;
        const float* xp = x + (size_t)nimg * (CCH * HWSZ) + hw;

        // ---- Load 32 channels for this pixel (coalesced per channel) -------
        float v[CCH];
#pragma unroll
        for (int c = 0; c < CCH; ++c) v[c] = xp[c * HWSZ];

        // ---- Prefetch next chunk into L2 (global_prefetch_b8) --------------
        {
            const int nbase = base + nWaves * 32;
            if (nbase < totalPix) {
                const int pn    = nbase + lane;
                const int nimg2 = pn / HWSZ;
                const int hw2   = pn - nimg2 * HWSZ;
                const float* xn = x + (size_t)nimg2 * (CCH * HWSZ) + hw2;
#pragma unroll
                for (int c = 0; c < CCH; c += 8)
                    __builtin_prefetch((const void*)(xn + c * HWSZ), 0, 0);
            }
        }

        // ---- Per-group softmax + sorted top4 + mean -> 20-dim stat ---------
        float s[KDIM];
#pragma unroll
        for (int g = 0; g < 4; ++g) {
            float e[8];
#pragma unroll
            for (int j = 0; j < 8; ++j) e[j] = v[g * 8 + j];
            float mx = e[0];
#pragma unroll
            for (int j = 1; j < 8; ++j) mx = fmaxf(mx, e[j]);
            float sum = 0.0f;
#pragma unroll
            for (int j = 0; j < 8; ++j) { e[j] = __expf(e[j] - mx); sum += e[j]; }
            const float inv = __builtin_amdgcn_rcpf(sum);

            // 19-comparator descending sort network (8 inputs)
            CE(0,1) CE(2,3) CE(4,5) CE(6,7)
            CE(0,2) CE(1,3) CE(4,6) CE(5,7)
            CE(1,2) CE(5,6)
            CE(0,4) CE(1,5) CE(2,6) CE(3,7)
            CE(1,4) CE(3,6)
            CE(2,4) CE(3,5)
            CE(3,4)

            s[g * 5 + 0] = e[0] * inv;
            s[g * 5 + 1] = e[1] * inv;
            s[g * 5 + 2] = e[2] * inv;
            s[g * 5 + 3] = e[3] * inv;
            s[g * 5 + 4] = (e[0] + e[1] + e[2] + e[3]) * inv * 0.25f;
        }

        // ---- Pack stats to f16 pairs: pso[i] = {s[2i], s[2i+1]} ------------
        unsigned int pso[10];
#pragma unroll
        for (int i = 0; i < 10; ++i) {
            union { _Float16 h[2]; unsigned int u; } pk;
            pk.h[0] = (_Float16)s[2 * i];
            pk.h[1] = (_Float16)s[2 * i + 1];
            pso[i] = pk.u;
        }
        // Exchange packed stats with partner half-wave (pixel ^ 16)
        unsigned int psp[10];
#pragma unroll
        for (int i = 0; i < 10; ++i)
            psp[i] = (unsigned int)__shfl_xor((int)pso[i], 16, 32);

        // ---- Assemble A tiles (16-bit A 16x32 layout) directly from pairs --
        // lanes 0-15:  VGPR0-3 = K pairs {0..3}, VGPR4-5 = pairs {8,9} (K16..19)
        // lanes 16-31: VGPR0-3 = K pairs {4..7} (K8..15), rest zero pad
        A16 a0, a1;
#pragma unroll
        for (int i = 0; i < 8; ++i) {
            const unsigned int lo_o = (i < 4) ? pso[i]
                                   : (i == 4) ? pso[8]
                                   : (i == 5) ? pso[9] : 0u;
            const unsigned int lo_p = (i < 4) ? psp[i]
                                   : (i == 4) ? psp[8]
                                   : (i == 5) ? psp[9] : 0u;
            const unsigned int hi_o = (i < 4) ? pso[4 + i] : 0u;
            const unsigned int hi_p = (i < 4) ? psp[4 + i] : 0u;
            a0.d[i] = hiH ? hi_p : lo_o;   // rows = pixels base+0..15
            a1.d[i] = hiH ? hi_o : lo_p;   // rows = pixels base+16..31
        }

        // ---- First conv: 8x v_wmma_f32_16x16x32_f16 (bias in C) ------------
        v8f D0[4], D1[4];
#pragma unroll
        for (int t = 0; t < 4; ++t) {
            v8f cin;
#pragma unroll
            for (int i = 0; i < 8; ++i) cin[i] = b1v[t];
            D0[t] = __builtin_amdgcn_wmma_f32_16x16x32_f16(
                false, a0.v, false, B[t], (short)0, cin, false, false);
            D1[t] = __builtin_amdgcn_wmma_f32_16x16x32_f16(
                false, a1.v, false, B[t], (short)0, cin, false, false);
        }

        // ---- ReLU + second conv (64->1): DPP16 butterfly reductions --------
        float acc0[8], acc1[8];
#pragma unroll
        for (int rt = 0; rt < 2; ++rt) {
#pragma unroll
            for (int r = 0; r < 8; ++r) {
                float acc = 0.0f;
#pragma unroll
                for (int t = 0; t < 4; ++t) {
                    const float h = (rt == 0) ? D0[t][r] : D1[t][r];
                    acc += relu1(h) * w2v[t];   // single v_med3_f32 ReLU
                }
                acc = row_dpp_add<DPP_XOR1>(acc);
                acc = row_dpp_add<DPP_XOR2>(acc);
                acc = row_dpp_add<DPP_XOR4>(acc);
                acc = row_dpp_add<DPP_XOR8>(acc);
                if (rt == 0) acc0[r] = acc; else acc1[r] = acc;
            }
        }

        // ---- Route each pixel's value to its lane; 1 sigmoid, 1 store ------
        // Binary select tree: pick acc[laneM & 7] of this half.
        const bool b0 = (lane & 1) != 0;
        const bool b1b = (lane & 2) != 0;
        const bool b2b = (lane & 4) != 0;
        const bool b3b = (lane & 8) != 0;

        float s01 = b0 ? acc0[1] : acc0[0];
        float s23 = b0 ? acc0[3] : acc0[2];
        float s45 = b0 ? acc0[5] : acc0[4];
        float s67 = b0 ? acc0[7] : acc0[6];
        float s03 = b1b ? s23 : s01;
        float s47 = b1b ? s67 : s45;
        const float sel0 = b2b ? s47 : s03;

        s01 = b0 ? acc1[1] : acc1[0];
        s23 = b0 ? acc1[3] : acc1[2];
        s45 = b0 ? acc1[5] : acc1[4];
        s67 = b0 ? acc1[7] : acc1[6];
        s03 = b1b ? s23 : s01;
        s47 = b1b ? s67 : s45;
        const float sel1 = b2b ? s47 : s03;

        // Rows 8-15 of tile0 live in the high half; rows 0-7 of tile1 in the
        // low half -> fetch the partner half's selection.
        const float oth0 = __shfl_xor(sel0, 16, 32);
        const float oth1 = __shfl_xor(sel1, 16, 32);

        // lanes 0-7: sel0 | 8-15: oth0 | 16-23: oth1 | 24-31: sel1
        const float fin = hiH ? (b3b ? sel1 : oth1)
                              : (b3b ? oth0 : sel0);

        const float z = fin + bias2;
        const float y = __builtin_amdgcn_rcpf(1.0f + __expf(-z));
        out[base + lane] = y;   // one coalesced 128B store per wave
    }
}

extern "C" void kernel_launch(void* const* d_in, const int* in_sizes, int n_in,
                              void* d_out, int out_size, void* d_ws, size_t ws_size,
                              hipStream_t stream) {
    const float* x  = (const float*)d_in[0];
    const float* w1 = (const float*)d_in[1];
    const float* b1 = (const float*)d_in[2];
    const float* w2 = (const float*)d_in[3];
    const float* b2 = (const float*)d_in[4];
    float* out = (float*)d_out;
    const int totalPix = out_size;  // 32*160*160 = 819200

    dgqp_kernel<<<1024, 256, 0, stream>>>(x, w1, b1, w2, b2, out, totalPix);
}
